// MSDeformAttnTransformerEncoderOnly_77103252898446
// MI455X (gfx1250) — matrix operand order repaired and checked by reading, not compiled, not run on hardware
//
#include <hip/hip_runtime.h>
#include <hip/hip_bf16.h>

// ---------------- problem constants ----------------
#define D_MODEL 256
#define NHEADS  8
#define HDIM    32
#define NLVL    4
#define NPTS    4
#define DFF     1024
#define LAYERS  6
#define LEN     3060          // 48*48 + 24*24 + 12*12 + 6*6
#define BATCH   2
#define MROWS   (BATCH*LEN)   // 6120

typedef __attribute__((ext_vector_type(16))) _Float16 v16h;
typedef __attribute__((ext_vector_type(16))) float    v16f;
typedef __attribute__((ext_vector_type(8)))  float    v8f;

typedef __attribute__((ext_vector_type(4))) unsigned int u32x4;
typedef __attribute__((ext_vector_type(8))) int          i32x8;
typedef __attribute__((ext_vector_type(4))) int          i32x4;

#if defined(__gfx1250__) && __has_builtin(__builtin_amdgcn_tensor_load_to_lds) && \
    __has_builtin(__builtin_amdgcn_s_wait_tensorcnt)
#define USE_TDM 1
#else
#define USE_TDM 0
#endif

// ============================================================
// Weight pre-pack: fp32 row-major (K,N)  ->  fp16 [K/32][N][32]
// so each lane's 16 contiguous K-values are two b128 loads
// (B k order used here, same as A: lanes 0-15 = K 0..15, lanes 16-31 = K 16..31
// within each 32-K block), and each 64-col x 32-K tile is one
// CONTIGUOUS 4KB chunk -> perfect for a 1-row TDM tile.
// ============================================================
__global__ __launch_bounds__(256)
void pack_weights_kernel(const float* __restrict__ W, _Float16* __restrict__ dst,
                         int K, int N)
{
    int idx = blockIdx.x * 256 + threadIdx.x;
    if (idx >= K * N) return;
    int n = idx % N;
    int k = idx / N;
    dst[((size_t)(k >> 5) * N + n) * 32 + (k & 31)] = (_Float16)W[(size_t)k * N + n];
}

#if USE_TDM
// Issue a Tensor-Data-Mover load of one contiguous 4KB (2048 x f16)
// B tile into LDS.  D# built per CDNA5 ISA.
__device__ __forceinline__ void tdm_load_b_tile(const _Float16* gsrc, unsigned int ldsByteOff)
{
    unsigned long long ga = (unsigned long long)(uintptr_t)gsrc;
    u32x4 g0;
    g0[0] = 1u;                                   // count=1, is_restore=0, gather=0
    g0[1] = ldsByteOff;                           // lds_addr (bytes)
    g0[2] = (unsigned int)ga;                     // global_addr[31:0]
    g0[3] = (unsigned int)((ga >> 32) & 0x1FFFFFFu) | (2u << 30); // addr[56:32] | type=2
    i32x8 g1;
    g1[0] = 0x00010000;                           // workgroup_mask=0, data_size=1 (2 bytes)
    g1[1] = (2048 & 0xFFFF) << 16;                // tensor_dim0 = 2048 (lo16 in [31:16])
    g1[2] = 0x00010000;                           // tensor_dim0 hi=0, tensor_dim1 = 1
    g1[3] = (2048 & 0xFFFF) << 16;                // tensor_dim1 hi=0, tile_dim0 = 2048
    g1[4] = 1;                                    // tile_dim1 = 1, tile_dim2 = 0
    g1[5] = 2048;                                 // tensor_dim0_stride = 2048
    g1[6] = 0;                                    // stride hi, tensor_dim1_stride lo
    g1[7] = 0;                                    // tensor_dim1_stride hi
    i32x4 z4 = {0, 0, 0, 0};
#if __clang_major__ >= 23
    i32x8 z8 = {0, 0, 0, 0, 0, 0, 0, 0};
    __builtin_amdgcn_tensor_load_to_lds(g0, g1, z4, z4, z8, 0);
#else
    __builtin_amdgcn_tensor_load_to_lds(g0, g1, z4, z4, 0);
#endif
}
#endif

// ============================================================
// WMMA GEMM:  C[M,N] = A[M,K](f32) @ Bpacked[K,N](f16) + bias
// block = 128 threads (4 waves), tile = 64x64, K-step = 32.
// B tiles are TDM-DMA'd into double-buffered LDS (wave 0 issues,
// s_wait_tensorcnt + barrier publishes); A fragments are built
// in registers from f32 global loads with f16 conversion.
// ============================================================
__global__ __launch_bounds__(128)
void gemm_wmma_kernel(const float* __restrict__ A, const _Float16* __restrict__ Bp,
                      const float* __restrict__ bias, float* __restrict__ C,
                      int M, int N, int K, int relu)
{
    const int lane  = threadIdx.x & 31;
    const int wave  = threadIdx.x >> 5;
    const int khalf = lane >> 4;      // 0: lanes 0-15, 1: lanes 16-31
    const int l15   = lane & 15;

    const int rowBase = blockIdx.y * 64 + wave * 16;
    const int colBase = blockIdx.x * 64;

    v8f acc[4] = {};

    // A fragment row for this lane (clamped; overhang rows masked at store)
    int arow = rowBase + l15;
    if (arow > M - 1) arow = M - 1;
    const float* Aptr = A + (size_t)arow * K;

    const int nKB = K >> 5;

#if USE_TDM
    __shared__ _Float16 Bs[2][2048];   // two 4KB B-tile buffers
    const unsigned int ldsOff[2] = {
        (unsigned int)(uintptr_t)&Bs[0][0],
        (unsigned int)(uintptr_t)&Bs[1][0]
    };
    // prologue: DMA tile kb=0
    if (wave == 0) {
        tdm_load_b_tile(Bp + (size_t)colBase * 32, ldsOff[0]);
        __builtin_amdgcn_s_wait_tensorcnt(0);
    }
    __syncthreads();
#endif

    for (int kb = 0; kb < nKB; ++kb) {
#if USE_TDM
        if (wave == 0 && kb + 1 < nKB)
            tdm_load_b_tile(Bp + ((size_t)(kb + 1) * N + colBase) * 32,
                            ldsOff[(kb + 1) & 1]);
#endif
        const int k0 = (kb << 5) + khalf * 16;  // A layout: khalf selects K{+0..15}/{+16..31}
        float4 f0 = *(const float4*)(Aptr + k0);
        float4 f1 = *(const float4*)(Aptr + k0 + 4);
        float4 f2 = *(const float4*)(Aptr + k0 + 8);
        float4 f3 = *(const float4*)(Aptr + k0 + 12);
        if (kb + 1 < nKB) __builtin_prefetch(Aptr + k0 + 32);   // global_prefetch_b8

        v16f af;
        af[0]=f0.x;  af[1]=f0.y;  af[2]=f0.z;  af[3]=f0.w;
        af[4]=f1.x;  af[5]=f1.y;  af[6]=f1.z;  af[7]=f1.w;
        af[8]=f2.x;  af[9]=f2.y;  af[10]=f2.z; af[11]=f2.w;
        af[12]=f3.x; af[13]=f3.y; af[14]=f3.z; af[15]=f3.w;
        v16h a = __builtin_convertvector(af, v16h);

        #pragma unroll
        for (int sub = 0; sub < 4; ++sub) {
            v16h b;
#if USE_TDM
            const _Float16* bp = &Bs[kb & 1][(sub * 16 + l15) * 32 + (khalf << 4)];
#else
            const int col = colBase + sub * 16 + l15;
            const _Float16* bp = Bp + ((size_t)kb * N + col) * 32 + (khalf << 4);
#endif
            ((uint4*)&b)[0] = *(const uint4*)(bp);
            ((uint4*)&b)[1] = *(const uint4*)(bp + 8);
            acc[sub] = __builtin_amdgcn_wmma_f32_16x16x32_f16(
                false, a, false, b, (short)0, acc[sub], false, false);
        }

#if USE_TDM
        if (kb + 1 < nKB) {
            if (wave == 0) __builtin_amdgcn_s_wait_tensorcnt(0);
            __syncthreads();
        }
#endif
    }

    #pragma unroll
    for (int sub = 0; sub < 4; ++sub) {
        const int col = colBase + sub * 16 + l15;
        const float bv = bias ? bias[col] : 0.0f;
        #pragma unroll
        for (int r = 0; r < 8; ++r) {
            const int row = rowBase + r + khalf * 8;   // C layout: vgpr r -> M=r (+8 hi lanes)
            if (row < M) {
                float v = acc[sub][r] + bv;
                if (relu) v = fmaxf(v, 0.0f);
                C[(size_t)row * N + col] = v;
            }
        }
    }
}

// ============================================================
// Flatten (bs,D,H,W) levels -> (bs,Len,D), pos += level_embed
// ============================================================
__device__ __forceinline__ void level_of(int l, int& lv, int& start, int& HW)
{
    if (l < 2304)      { lv = 0; start = 0;    HW = 48; }
    else if (l < 2880) { lv = 1; start = 2304; HW = 24; }
    else if (l < 3024) { lv = 2; start = 2880; HW = 12; }
    else               { lv = 3; start = 3024; HW = 6;  }
}

__global__ __launch_bounds__(256)
void flatten_kernel(const float* __restrict__ s0, const float* __restrict__ s1,
                    const float* __restrict__ s2, const float* __restrict__ s3,
                    const float* __restrict__ p0, const float* __restrict__ p1,
                    const float* __restrict__ p2, const float* __restrict__ p3,
                    const float* __restrict__ lvlEmb,
                    float* __restrict__ srcFlat, float* __restrict__ posFlat)
{
    int idx = blockIdx.x * 256 + threadIdx.x;
    if (idx >= BATCH * LEN * D_MODEL) return;
    int d = idx & 255;
    int l = (idx >> 8) % LEN;
    int b = idx / (LEN * D_MODEL);
    int lv, start, HW;
    level_of(l, lv, start, HW);
    int p = l - start;
    size_t off = ((size_t)(b * D_MODEL + d)) * (HW * HW) + p;
    const float* sp; const float* pp;
    if (lv == 0)      { sp = s0; pp = p0; }
    else if (lv == 1) { sp = s1; pp = p1; }
    else if (lv == 2) { sp = s2; pp = p2; }
    else              { sp = s3; pp = p3; }
    srcFlat[idx] = sp[off];
    posFlat[idx] = pp[off] + lvlEmb[lv * D_MODEL + d];
}

__global__ __launch_bounds__(256)
void ref_kernel(float* __restrict__ ref)
{
    int l = blockIdx.x * 256 + threadIdx.x;
    if (l >= LEN) return;
    int lv, start, HW;
    level_of(l, lv, start, HW);
    int p = l - start;
    int y = p / HW, x = p % HW;
    ref[l * 2 + 0] = (x + 0.5f) / HW;
    ref[l * 2 + 1] = (y + 0.5f) / HW;
}

__global__ __launch_bounds__(256)
void add_kernel(const float* __restrict__ a, const float* __restrict__ b,
                float* __restrict__ c, int n)
{
    int i = blockIdx.x * 256 + threadIdx.x;
    if (i < n) c[i] = a[i] + b[i];
}

// softmax over the last-16 group of attn logits (bs,Len,NH,16)
__global__ __launch_bounds__(256)
void softmax16_kernel(float* __restrict__ attn, int total)
{
    int i = blockIdx.x * 256 + threadIdx.x;
    if (i >= total) return;
    float* p = attn + (size_t)i * 16;
    float m = p[0];
    #pragma unroll
    for (int j = 1; j < 16; ++j) m = fmaxf(m, p[j]);
    float s = 0.0f;
    float e[16];
    #pragma unroll
    for (int j = 0; j < 16; ++j) { e[j] = __expf(p[j] - m); s += e[j]; }
    float inv = 1.0f / s;
    #pragma unroll
    for (int j = 0; j < 16; ++j) p[j] = e[j] * inv;
}

// ============================================================
// MSDA sampling: one wave per (b, query, head); lane = channel.
// ============================================================
__global__ __launch_bounds__(256)
void msda_kernel(const float* __restrict__ value, const float* __restrict__ off,
                 const float* __restrict__ attn, const float* __restrict__ ref,
                 float* __restrict__ out)
{
    const int wid  = (blockIdx.x * 256 + threadIdx.x) >> 5;
    const int lane = threadIdx.x & 31;
    if (wid >= BATCH * LEN * NHEADS) return;
    const int h = wid & 7;
    const int l = (wid >> 3) % LEN;
    const int b = wid / (NHEADS * LEN);

    const float rx = ref[l * 2 + 0];
    const float ry = ref[l * 2 + 1];
    const size_t rowOff = (size_t)(b * LEN + l) * D_MODEL;
    const float* offp = off + rowOff + h * 32;                       // (h,lv,p,2)
    const float* attp = attn + (size_t)(b * LEN + l) * 128 + h * 16; // (h,lv,p)
    const float* vbase = value + (size_t)b * LEN * D_MODEL + h * HDIM + lane;

    const int HWs[4]    = {48, 24, 12, 6};
    const int starts[4] = {0, 2304, 2880, 3024};

    float acc = 0.0f;
    #pragma unroll
    for (int lv = 0; lv < NLVL; ++lv) {
        const int HW = HWs[lv];
        const int st = starts[lv];
        #pragma unroll
        for (int p = 0; p < NPTS; ++p) {
            float ox = offp[lv * 8 + p * 2 + 0];
            float oy = offp[lv * 8 + p * 2 + 1];
            float aw = attp[lv * 4 + p];
            float x = rx * HW + ox - 0.5f;
            float y = ry * HW + oy - 0.5f;
            float x0f = floorf(x), y0f = floorf(y);
            float lx = x - x0f, ly = y - y0f;
            int x0 = (int)x0f, y0 = (int)y0f;
            float s = 0.0f;
            #pragma unroll
            for (int dy = 0; dy < 2; ++dy) {
                #pragma unroll
                for (int dx = 0; dx < 2; ++dx) {
                    int xi = x0 + dx, yi = y0 + dy;
                    if (xi >= 0 && xi < HW && yi >= 0 && yi < HW) {
                        float w = (dx ? lx : 1.0f - lx) * (dy ? ly : 1.0f - ly);
                        s += w * vbase[(size_t)(st + yi * HW + xi) * D_MODEL];
                    }
                }
            }
            acc += aw * s;
        }
    }
    out[rowOff + h * HDIM + lane] = acc;
}

// ============================================================
// Residual + LayerNorm (in place on x): x = LN(x + res)*g + beta
// one row (256 channels) per 256-thread block; LDS reduction.
// ============================================================
__global__ __launch_bounds__(256)
void ln_res_kernel(float* __restrict__ x, const float* __restrict__ res,
                   const float* __restrict__ g, const float* __restrict__ beta)
{
    __shared__ float red[256];
    const int row = blockIdx.x;
    const int t = threadIdx.x;
    const size_t idx = (size_t)row * D_MODEL + t;
    float v = x[idx] + res[idx];
    red[t] = v;
    __syncthreads();
    #pragma unroll
    for (int s = 128; s > 0; s >>= 1) {
        if (t < s) red[t] += red[t + s];
        __syncthreads();
    }
    float mean = red[0] * (1.0f / D_MODEL);
    __syncthreads();
    float d = v - mean;
    red[t] = d * d;
    __syncthreads();
    #pragma unroll
    for (int s = 128; s > 0; s >>= 1) {
        if (t < s) red[t] += red[t + s];
        __syncthreads();
    }
    float var = red[0] * (1.0f / D_MODEL);
    x[idx] = d * rsqrtf(var + 1e-5f) * g[t] + beta[t];
}

__global__ void tail_kernel(int* __restrict__ t)
{
    if (threadIdx.x == 0 && blockIdx.x == 0) {
        const int v[12] = {48, 48, 24, 24, 12, 12, 6, 6, 0, 2304, 2880, 3024};
        #pragma unroll
        for (int i = 0; i < 12; ++i) t[i] = v[i];
    }
}

// ============================================================
// host launch
// ============================================================
extern "C" void kernel_launch(void* const* d_in, const int* in_sizes, int n_in,
                              void* d_out, int out_size, void* d_ws, size_t ws_size,
                              hipStream_t stream)
{
    (void)in_sizes; (void)n_in; (void)out_size; (void)ws_size;

    // input order: src0,pos0,src1,pos1,src2,pos2,src3,pos3, level_embed,
    //              Wv,bv,Woff,boff,Wa,ba,Wo,bo, g1,beta1,W1,bf1,W2,bf2,g2,beta2
    const float* src[4] = {(const float*)d_in[0], (const float*)d_in[2],
                           (const float*)d_in[4], (const float*)d_in[6]};
    const float* pos[4] = {(const float*)d_in[1], (const float*)d_in[3],
                           (const float*)d_in[5], (const float*)d_in[7]};
    const float* lvlEmb = (const float*)d_in[8];
    const float* Wv   = (const float*)d_in[9];
    const float* bv   = (const float*)d_in[10];
    const float* Woff = (const float*)d_in[11];
    const float* boff = (const float*)d_in[12];
    const float* Wa   = (const float*)d_in[13];
    const float* ba   = (const float*)d_in[14];
    const float* Wo   = (const float*)d_in[15];
    const float* bo   = (const float*)d_in[16];
    const float* g1   = (const float*)d_in[17];
    const float* be1  = (const float*)d_in[18];
    const float* W1   = (const float*)d_in[19];
    const float* bf1  = (const float*)d_in[20];
    const float* W2   = (const float*)d_in[21];
    const float* bf2  = (const float*)d_in[22];
    const float* g2   = (const float*)d_in[23];
    const float* be2  = (const float*)d_in[24];

    // ---- workspace carve ----
    char* ws = (char*)d_ws;
    size_t o = 0;
    auto carve = [&](size_t bytes) -> char* {
        char* r = ws + o;
        o += (bytes + 255) & ~(size_t)255;
        return r;
    };
    const size_t PACK_PER_LAYER = 753664;            // halfs per layer
    _Float16* wpack = (_Float16*)carve(LAYERS * PACK_PER_LAYER * sizeof(_Float16));
    float* bufX  = (float*)carve((size_t)MROWS * D_MODEL * 4);   // running "out"
    float* posF  = (float*)carve((size_t)MROWS * D_MODEL * 4);
    float* refB  = (float*)carve((size_t)LEN * 2 * 4);
    float* qB    = (float*)carve((size_t)MROWS * D_MODEL * 4);
    float* valB  = (float*)carve((size_t)MROWS * D_MODEL * 4);
    float* offB  = (float*)carve((size_t)MROWS * D_MODEL * 4);
    float* attnB = (float*)carve((size_t)MROWS * 128 * 4);
    float* msdaB = (float*)carve((size_t)MROWS * D_MODEL * 4);
    float* projB = (float*)carve((size_t)MROWS * D_MODEL * 4);
    float* hB    = (float*)carve((size_t)MROWS * DFF * 4);
    float* h2B   = (float*)carve((size_t)MROWS * D_MODEL * 4);

    // ---- pack all weights to fp16 WMMA-friendly layout ----
    struct Mat { const float* W; int K; int N; size_t off; };
    const Mat mats[6] = {
        {Wv,   256, 256,  0},
        {Woff, 256, 256,  65536},
        {Wa,   256, 128,  131072},
        {Wo,   256, 256,  163840},
        {W1,   256, 1024, 229376},
        {W2,   1024, 256, 491520},
    };
    for (int l = 0; l < LAYERS; ++l) {
        for (int m = 0; m < 6; ++m) {
            int total = mats[m].K * mats[m].N;
            pack_weights_kernel<<<(total + 255) / 256, 256, 0, stream>>>(
                mats[m].W + (size_t)l * total,
                wpack + (size_t)l * PACK_PER_LAYER + mats[m].off,
                mats[m].K, mats[m].N);
        }
    }

    // ---- flatten inputs & reference points ----
    flatten_kernel<<<MROWS, 256, 0, stream>>>(
        src[0], src[1], src[2], src[3], pos[0], pos[1], pos[2], pos[3],
        lvlEmb, bufX, posF);
    ref_kernel<<<(LEN + 255) / 256, 256, 0, stream>>>(refB);

    const dim3 gN256(4, 96), gN128(2, 96), gN1024(16, 96);
    const int nElem = MROWS * D_MODEL;

    for (int l = 0; l < LAYERS; ++l) {
        const _Float16* wp = wpack + (size_t)l * PACK_PER_LAYER;

        // q = out + pos
        add_kernel<<<MROWS, 256, 0, stream>>>(bufX, posF, qB, nElem);

        // value = out @ Wv + bv
        gemm_wmma_kernel<<<gN256, 128, 0, stream>>>(
            bufX, wp + 0, bv + l * 256, valB, MROWS, 256, 256, 0);
        // off = q @ Woff + boff
        gemm_wmma_kernel<<<gN256, 128, 0, stream>>>(
            qB, wp + 65536, boff + l * 256, offB, MROWS, 256, 256, 0);
        // attn logits = q @ Wa + ba
        gemm_wmma_kernel<<<gN128, 128, 0, stream>>>(
            qB, wp + 131072, ba + l * 128, attnB, MROWS, 128, 256, 0);

        softmax16_kernel<<<(MROWS * NHEADS + 255) / 256, 256, 0, stream>>>(
            attnB, MROWS * NHEADS);

        msda_kernel<<<MROWS, 256, 0, stream>>>(valB, offB, attnB, refB, msdaB);

        // out2 = msda @ Wo + bo
        gemm_wmma_kernel<<<gN256, 128, 0, stream>>>(
            msdaB, wp + 163840, bo + l * 256, projB, MROWS, 256, 256, 0);

        // out = LN(out + out2)
        ln_res_kernel<<<MROWS, 256, 0, stream>>>(bufX, projB, g1 + l * 256, be1 + l * 256);

        // h = relu(out @ W1 + bf1)
        gemm_wmma_kernel<<<gN1024, 128, 0, stream>>>(
            bufX, wp + 229376, bf1 + l * 1024, hB, MROWS, 1024, 256, 1);
        // h2 = h @ W2 + bf2
        gemm_wmma_kernel<<<gN256, 128, 0, stream>>>(
            hB, wp + 491520, bf2 + l * 256, h2B, MROWS, 256, 1024, 0);

        // out = LN(out + h2)
        ln_res_kernel<<<MROWS, 256, 0, stream>>>(bufX, h2B, g2 + l * 256, be2 + l * 256);
    }

    // memory -> d_out, then the two int32 tail outputs
    hipMemcpyAsync(d_out, bufX, (size_t)MROWS * D_MODEL * sizeof(float),
                   hipMemcpyDeviceToDevice, stream);
    tail_kernel<<<1, 32, 0, stream>>>(((int*)d_out) + (size_t)MROWS * D_MODEL);
}
